// DeformConv2d_29051158790803
// MI455X (gfx1250) — matrix operand order repaired and checked by reading, not compiled, not run on hardware
//
#include <hip/hip_runtime.h>

typedef __attribute__((ext_vector_type(16))) _Float16 v16h;
typedef __attribute__((ext_vector_type(8)))  _Float16 v8h;
typedef __attribute__((ext_vector_type(8)))  float    v8f;

#define NB    8
#define CIN   64
#define HH    128
#define WW    128
#define COUT  64
#define K2    9
#define HO    128
#define WO    128

// ---------------------------------------------------------------------------
// Kernel 1: NCHW f32 -> NHWC f16 transpose of x (LDS tile transpose).
// f16 staging halves both the gather bytes and the gather instruction count
// in the main kernel (8 contiguous channels per b128 instead of 4).
// ---------------------------------------------------------------------------
__global__ void __launch_bounds__(256)
xpose_kernel(const float* __restrict__ x, _Float16* __restrict__ xt) {
  __shared__ float tile[32][33];
  const int bx = blockIdx.x;          // W tile   (0..3)
  const int by = blockIdx.y;          // C tile   (0..1)
  const int bz = blockIdx.z;          // b*H + y  (0..1023)
  const int tx = threadIdx.x;         // 0..31
  const int ty = threadIdx.y;         // 0..7
  const int b  = bz >> 7;
  const int y  = bz & 127;
#pragma unroll
  for (int i = 0; i < 4; ++i) {
    const int c  = by * 32 + ty + i * 8;
    const int xc = bx * 32 + tx;
    tile[ty + i * 8][tx] = x[((b * CIN + c) * HH + y) * WW + xc];  // coalesced along W
  }
  __syncthreads();
#pragma unroll
  for (int i = 0; i < 4; ++i) {
    const int xc = bx * 32 + ty + i * 8;
    const int c  = by * 32 + tx;
    xt[((b * HH + y) * WW + xc) * CIN + c] = (_Float16)tile[tx][ty + i * 8];
  }
}

// ---------------------------------------------------------------------------
// Kernel 2: pack weight (COUT, CIN, 9) into WMMA B-fragment order.
// Global K index = kpoint*64 + c.  Per 16x16x32 step kk (18 steps) and
// N-tile nt (4 tiles): lane l holds column N = nt*16 + (l&15); element j of
// the v16h fragment holds K32 = j + (l>=16 ? 16 : 0).
// ---------------------------------------------------------------------------
__global__ void __launch_bounds__(256)
wpack_kernel(const float* __restrict__ w, _Float16* __restrict__ wB) {
  const int tid = blockIdx.x * 256 + threadIdx.x;   // 0..36863
  const int j   = tid & 15;
  const int l   = (tid >> 4) & 31;
  const int nt  = (tid >> 9) & 3;
  const int kk  = tid >> 11;                        // 0..17
  const int K32 = j + ((l & 16) ? 16 : 0);
  const int kpt = kk >> 1;
  const int c   = ((kk & 1) << 5) + K32;
  const int o   = (nt << 4) + (l & 15);
  wB[tid] = (_Float16)w[(o * CIN + c) * K2 + kpt];
}

// ---------------------------------------------------------------------------
// Kernel 3: main deformable conv. One wave per 16(pos) x 64(cout) tile.
// ---------------------------------------------------------------------------
__global__ void __launch_bounds__(256)
deform_main(const _Float16* __restrict__ xt, const float* __restrict__ off,
            const _Float16* __restrict__ wB, float* __restrict__ out) {
  __shared__ float ldsC[8][COUT * 16];              // 8 waves * 4 KB = 32 KB

  const int wave = threadIdx.x >> 5;
  const int lane = threadIdx.x & 31;
  const int tile = blockIdx.x * 8 + wave;           // 0..8191
  const int b    = tile >> 10;
  const int idx  = tile & 1023;
  const int ho   = idx >> 3;
  const int wo0  = (idx & 7) << 4;
  const int m    = lane & 15;                       // A-matrix row = output position
  const int hi   = lane >> 4;                       // lane half selects K sub-lanes
  const int wo   = wo0 + m;

  v8f acc[4] = {};                                  // 4 N-tiles of f32 C/D

  for (int k = 0; k < K2; ++k) {
    const int ky = k / 3, kx = k % 3;
    const float dy = off[((b * 18 + 2 * k    ) * HO + ho) * WO + wo];
    const float dx = off[((b * 18 + 2 * k + 1) * HO + ho) * WO + wo];
    const float py = dy + (float)(ky + ho - 1);
    const float px = dx + (float)(kx + wo - 1);
    const float y0f = floorf(py), x0f = floorf(px);
    const int   y0  = (int)y0f,   x0  = (int)x0f;
    const float wy1 = py - y0f,   wx1 = px - x0f;
    const float wy0 = 1.f - wy1,  wx0 = 1.f - wx1;

    const _Float16* p[4];
    _Float16 cwh[4];
#pragma unroll
    for (int cy = 0; cy < 2; ++cy) {
#pragma unroll
      for (int cx = 0; cx < 2; ++cx) {
        const int yy = y0 + cy, xx = x0 + cx;
        const bool v = (yy >= 0) && (yy < HH) && (xx >= 0) && (xx < WW);
        int yc = yy < 0 ? 0 : (yy > HH - 1 ? HH - 1 : yy);
        int xc = xx < 0 ? 0 : (xx > WW - 1 ? WW - 1 : xx);
        p[cy * 2 + cx]   = xt + (size_t)(((b * HH + yc) * WW + xc) * CIN);
        const float wgt  = (cy ? wy1 : wy0) * (cx ? wx1 : wx0);
        cwh[cy * 2 + cx] = (_Float16)(v ? wgt : 0.f);   // select, no divergence
      }
    }

#pragma unroll
    for (int half = 0; half < 2; ++half) {
      const int kk = 2 * k + half;

      // Issue all 4 B-fragment loads FIRST: no deps on this iteration, so
      // their L2 latency hides behind the 8 gather loads + 32 pk_fma blends
      // below instead of stalling each WMMA on s_wait_loadcnt 0x0.
      v16h bfs[4];
#pragma unroll
      for (int nt = 0; nt < 4; ++nt) {
        bfs[nt] = *(const v16h*)(wB + (((kk * 4 + nt) * 32 + lane) << 4));
      }

      // A fragment: lane holds M=m; element j covers channel
      //   c = half*32 + hi*16 + j   (contiguous k order, same as wB packing)
      const int cbase = (half << 5) + (hi << 4);
      v8h s0 = {};                                  // channels cbase..cbase+7
      v8h s1 = {};                                  // channels cbase+8..cbase+15
#pragma unroll
      for (int q = 0; q < 4; ++q) {                 // 4 bilinear corners
        const v8h g0 = *(const v8h*)(p[q] + cbase);        // one b128 each
        const v8h g1 = *(const v8h*)(p[q] + cbase + 8);
        const _Float16 cq = cwh[q];
        const v8h cs = {cq, cq, cq, cq, cq, cq, cq, cq};
        s0 += cs * g0;                              // v_pk_fma_f16
        s1 += cs * g1;
      }
      v16h a;
#pragma unroll
      for (int e = 0; e < 8; ++e) { a[e] = s0[e]; a[e + 8] = s1[e]; }

#pragma unroll
      for (int nt = 0; nt < 4; ++nt) {
        acc[nt] = __builtin_amdgcn_wmma_f32_16x16x32_f16(
            false, a, false, bfs[nt], (short)0, acc[nt], false, false);
      }
    }
  }

  // Stage D tiles through LDS so NCHW stores are 64B-coalesced per cout row.
  // D layout: lane half selects M rows 0-7 / 8-15, VGPR r = M%8, N = lane%16.
#pragma unroll
  for (int nt = 0; nt < 4; ++nt) {
    const int o = (nt << 4) + m;
#pragma unroll
    for (int r = 0; r < 8; ++r) {
      ldsC[wave][o * 16 + (r + (hi << 3))] = acc[nt][r];
    }
  }
  __syncthreads();
#pragma unroll 4
  for (int i = 0; i < 32; ++i) {
    const int e  = i * 32 + lane;
    const int o  = e >> 4;
    const int mm = e & 15;
    out[((b * COUT + o) * HO + ho) * WO + wo0 + mm] = ldsC[wave][e];
  }
}

// ---------------------------------------------------------------------------
extern "C" void kernel_launch(void* const* d_in, const int* in_sizes, int n_in,
                              void* d_out, int out_size, void* d_ws, size_t ws_size,
                              hipStream_t stream) {
  const float* x   = (const float*)d_in[0];
  const float* off = (const float*)d_in[1];
  const float* w   = (const float*)d_in[2];
  float* out = (float*)d_out;

  _Float16* xt = (_Float16*)d_ws;
  _Float16* wB = (_Float16*)((char*)d_ws + (size_t)NB * CIN * HH * WW * sizeof(_Float16));

  dim3 tb(32, 8, 1);
  dim3 tg(WW / 32, CIN / 32, NB * HH);
  xpose_kernel<<<tg, tb, 0, stream>>>(x, xt);

  wpack_kernel<<<(18 * 4 * 32 * 16) / 256, 256, 0, stream>>>(w, wB);

  deform_main<<<(NB * HO * WO / 16) / 8, 256, 0, stream>>>(xt, off, wB, out);
}
